// AttentiveGLA_36146444763875
// MI455X (gfx1250) — matrix operand order repaired and checked by reading, not compiled, not run on hardware
//
#include <hip/hip_runtime.h>

// ---------------------------------------------------------------------------
// Model constants (from reference)
// ---------------------------------------------------------------------------
namespace {
constexpr int Dm  = 1024;
constexpr int DCc = 512;
constexpr int Hh  = 32;
constexpr int Bc  = 2;
constexpr int Nt  = 256;
constexpr int Jt  = 256;
constexpr int DKk = Dm / Hh;        // 32
constexpr int DVv = 2 * Dm / Hh;    // 64
constexpr int HINc = Dm * 4 / 3;    // 1365
constexpr int HSW_LD = 1368;        // hsw row stride padded to multiple of 4
constexpr int BN  = Bc * Nt;        // 512
}

typedef __attribute__((ext_vector_type(16))) __bf16 v16bf;
typedef __attribute__((ext_vector_type(8)))  float  v8f;
typedef __attribute__((ext_vector_type(4)))  float  f4;

#define ACT_NONE 0
#define ACT_LOGSIG_GLN 1

// ---------------------------------------------------------------------------
// Shared epilogue: C/D layout VGPR r -> M = row0 + r + 8*laneHi, N = ncol
// ---------------------------------------------------------------------------
__device__ __forceinline__ void gemm_epilogue(
    const v8f& acc, float* __restrict__ C, const float* __restrict__ bias,
    const float* __restrict__ resid, int row0, int laneHi, int ncol, int N,
    int ldc, float alpha, int act)
{
    #pragma unroll
    for (int r = 0; r < 8; ++r) {
        const int row = row0 + r + (laneHi << 3);
        if (ncol < N) {
            float v = acc[r] * alpha;
            if (bias) v += bias[ncol];
            if (act == ACT_LOGSIG_GLN)
                v = (fminf(v, 0.0f) - log1pf(expf(-fabsf(v)))) * (1.0f / 16.0f);
            if (resid) v += resid[(long long)row * ldc + ncol];
            C[(long long)row * ldc + ncol] = v;
        }
    }
}

// ---------------------------------------------------------------------------
// Async staging of one 32(K) x 16(N) f32 B-tile into LDS, transposed to
// n-major ([n][k]) so fragment reads are contiguous.  Each of the 8 waves
// issues 2 GLOBAL_LOAD_ASYNC_TO_LDS_B32 ops (32 lanes x 4B each = 128B/op);
// each lane scatters one dword to its own LDS destination (ASYNCcnt-tracked).
// ---------------------------------------------------------------------------
__device__ __forceinline__ void async_stage_b(
    const float* __restrict__ Bm, int K, int N, int ldb, int col0, int k0,
    unsigned ldsOff, int wave, int lane)
{
    const int n = lane & 15;
    const int c = min(col0 + n, N - 1);
    #pragma unroll
    for (int j = 0; j < 2; ++j) {
        const int kl = (wave << 2) + (j << 1) + (lane >> 4);   // 0..31 over block
        const float* gp = Bm + (long long)min(k0 + kl, K - 1) * ldb + c;
        const unsigned loff = ldsOff + (unsigned)((n * 32 + kl) * 4);
        asm volatile("global_load_async_to_lds_b32 %0, %1, off"
                     :: "v"(loff), "v"(gp) : "memory");
    }
}

// ---------------------------------------------------------------------------
// NN GEMM: C = act(alpha*A(MxK)*B(KxN) + bias) [+resid].  A row-major,
// lda % 4 == 0 and M % 16 == 0 required.  B row-major, any ldb.
// All 8 waves in a block share one B tile: the tile is staged into LDS with
// async-to-LDS scatter (transposed), double buffered; lanes read 4x b128.
// ---------------------------------------------------------------------------
__global__ __launch_bounds__(256)
void gemm_nn_wmma(const float* __restrict__ A, const float* __restrict__ Bm,
                  const float* __restrict__ bias, const float* __restrict__ resid,
                  float* __restrict__ C,
                  int M, int N, int K, int lda, int ldb, int ldc,
                  long long strideA, long long strideB, long long strideC,
                  long long strideR, float alpha, int act)
{
    __shared__ __align__(64) float ldsBT[2][16][32];    // [buf][n][k], f32

    const int b = blockIdx.z;
    A  += (long long)b * strideA;
    Bm += (long long)b * strideB;
    C  += (long long)b * strideC;
    if (resid) resid += (long long)b * strideR;

    const int wave   = threadIdx.x >> 5;
    const int lane   = threadIdx.x & 31;
    const int laneLo = lane & 15;
    const int laneHi = lane >> 4;

    const int mtiles = (M + 15) >> 4;
    int tileM = blockIdx.y * 8 + wave;
    if (tileM >= mtiles) tileM = mtiles - 1;     // clamp: stay in barriers, EXEC full
    const int tileN = blockIdx.x;

    const int row0 = tileM << 4;
    const int col0 = tileN << 4;
    const int m    = row0 + laneLo;              // always < M (M % 16 == 0)
    const int ncol = col0 + laneLo;
    const float* ap = A + (long long)m * lda;

    const unsigned ldsBase = (unsigned)(unsigned long long)&ldsBT[0][0][0];

    v8f acc = {};
    async_stage_b(Bm, K, N, ldb, col0, 0, ldsBase, wave, lane);
    int buf = 0;
    for (int k0 = 0; k0 < K; k0 += 32) {
        asm volatile("s_wait_asynccnt 0x0" ::: "memory");  // own scatters done
        __syncthreads();                                   // whole tile visible
        if (k0 + 32 < K)
            async_stage_b(Bm, K, N, ldb, col0, k0 + 32,
                          ldsBase + (unsigned)((buf ^ 1) * 2048), wave, lane);

        // A fragment: lane holds M=laneLo, K = kbA+{0..15}
        const int kbA = k0 + (laneHi << 4);
        v16bf fa;
        if (k0 + 32 <= K) {                       // fast path: 4x b128
            const f4 x0 = *(const f4*)(ap + kbA);
            const f4 x1 = *(const f4*)(ap + kbA + 4);
            const f4 x2 = *(const f4*)(ap + kbA + 8);
            const f4 x3 = *(const f4*)(ap + kbA + 12);
            #pragma unroll
            for (int i = 0; i < 4; ++i) {
                fa[i]      = (__bf16)x0[i];
                fa[4 + i]  = (__bf16)x1[i];
                fa[8 + i]  = (__bf16)x2[i];
                fa[12 + i] = (__bf16)x3[i];
            }
        } else {                                  // ragged tail: clamp + select
            #pragma unroll
            for (int i = 0; i < 8; ++i) {
                const int k1 = kbA + i, k2 = kbA + 8 + i;
                float a0 = ap[min(k1, K - 1)]; a0 = (k1 < K) ? a0 : 0.0f;
                float a1 = ap[min(k2, K - 1)]; a1 = (k2 < K) ? a1 : 0.0f;
                fa[i]     = (__bf16)a0;
                fa[8 + i] = (__bf16)a1;
            }
        }

        // B fragment: lane holds N=laneLo, K = 16*laneHi + e  (contiguous in LDS)
        const float* lp = &ldsBT[buf][laneLo][laneHi << 4];
        v16bf fb;
        if (k0 + 32 <= K) {
            const f4 y0 = *(const f4*)(lp);
            const f4 y1 = *(const f4*)(lp + 4);
            const f4 y2 = *(const f4*)(lp + 8);
            const f4 y3 = *(const f4*)(lp + 12);
            #pragma unroll
            for (int i = 0; i < 4; ++i) {
                fb[i]      = (__bf16)y0[i];
                fb[4 + i]  = (__bf16)y1[i];
                fb[8 + i]  = (__bf16)y2[i];
                fb[12 + i] = (__bf16)y3[i];
            }
        } else {                                  // zero out K-overhang
            #pragma unroll
            for (int e = 0; e < 16; ++e) {
                const int kk = k0 + (laneHi << 4) + e;
                const float bv = lp[e];
                fb[e] = (__bf16)((kk < K) ? bv : 0.0f);
            }
        }

        acc = __builtin_amdgcn_wmma_f32_16x16x32_bf16(
                  false, fa, false, fb, (short)0, acc, false, false);
        buf ^= 1;
    }
    gemm_epilogue(acc, C, bias, resid, row0, laneHi, ncol, N, ldc, alpha, act);
}

// ---------------------------------------------------------------------------
// NT GEMM: element(k,n) = B[n*ldb + k]  (B rows contiguous in K -> direct
// vectorized loads, no LDS).  lda % 4 == 0, ldb % 4 == 0, M % 16 == 0.
// ---------------------------------------------------------------------------
__global__ __launch_bounds__(256)
void gemm_nt_wmma(const float* __restrict__ A, const float* __restrict__ Bm,
                  const float* __restrict__ bias, const float* __restrict__ resid,
                  float* __restrict__ C,
                  int M, int N, int K, int lda, int ldb, int ldc,
                  long long strideA, long long strideB, long long strideC,
                  long long strideR, float alpha, int act)
{
    const int b = blockIdx.z;
    A  += (long long)b * strideA;
    Bm += (long long)b * strideB;
    C  += (long long)b * strideC;
    if (resid) resid += (long long)b * strideR;

    const int wave   = threadIdx.x >> 5;
    const int lane   = threadIdx.x & 31;
    const int laneLo = lane & 15;
    const int laneHi = lane >> 4;

    const int mtiles = (M + 15) >> 4;
    int tileM = blockIdx.y * 8 + wave;
    if (tileM >= mtiles) tileM = mtiles - 1;
    const int tileN = blockIdx.x;

    const int row0 = tileM << 4;
    const int col0 = tileN << 4;
    const int m    = row0 + laneLo;
    const int ncol = col0 + laneLo;
    const float* ap = A + (long long)m * lda;
    const float* bp = Bm + (long long)min(ncol, N - 1) * ldb;

    v8f acc = {};
    for (int k0 = 0; k0 < K; k0 += 32) {
        const int kbA = k0 + (laneHi << 4);
        const int kbB = k0 + (laneHi << 4);
        v16bf fa, fb;
        if (k0 + 32 <= K) {
            const f4 x0 = *(const f4*)(ap + kbA);
            const f4 x1 = *(const f4*)(ap + kbA + 4);
            const f4 x2 = *(const f4*)(ap + kbA + 8);
            const f4 x3 = *(const f4*)(ap + kbA + 12);
            const f4 y0 = *(const f4*)(bp + kbB);
            const f4 y1 = *(const f4*)(bp + kbB + 4);
            const f4 y2 = *(const f4*)(bp + kbB + 8);
            const f4 y3 = *(const f4*)(bp + kbB + 12);
            #pragma unroll
            for (int i = 0; i < 4; ++i) {
                fa[i]      = (__bf16)x0[i];
                fa[4 + i]  = (__bf16)x1[i];
                fa[8 + i]  = (__bf16)x2[i];
                fa[12 + i] = (__bf16)x3[i];
                fb[i]      = (__bf16)y0[i];
                fb[4 + i]  = (__bf16)y1[i];
                fb[8 + i]  = (__bf16)y2[i];
                fb[12 + i] = (__bf16)y3[i];
            }
        } else {
            #pragma unroll
            for (int i = 0; i < 8; ++i) {
                const int k1 = kbA + i, k2 = kbA + 8 + i;
                float a0 = ap[min(k1, K - 1)]; a0 = (k1 < K) ? a0 : 0.0f;
                float a1 = ap[min(k2, K - 1)]; a1 = (k2 < K) ? a1 : 0.0f;
                fa[i]     = (__bf16)a0;
                fa[8 + i] = (__bf16)a1;
            }
            #pragma unroll
            for (int e = 0; e < 16; ++e) {
                const int kk = kbB + e;
                float bv = bp[min(kk, K - 1)]; bv = (kk < K) ? bv : 0.0f;
                fb[e] = (__bf16)bv;
            }
        }
        acc = __builtin_amdgcn_wmma_f32_16x16x32_bf16(
                  false, fa, false, fb, (short)0, acc, false, false);
    }
    gemm_epilogue(acc, C, bias, resid, row0, laneHi, ncol, N, ldc, alpha, act);
}

// ---------------------------------------------------------------------------
// LayerNorm over last dim (one block per row)
// ---------------------------------------------------------------------------
__global__ __launch_bounds__(256)
void layernorm_kernel(const float* __restrict__ x, const float* __restrict__ g,
                      const float* __restrict__ bta, float* __restrict__ out, int dim)
{
    __shared__ float red[256];
    const long long row = blockIdx.x;
    const float* xr = x + row * dim;
    float s = 0.0f;
    for (int i = threadIdx.x; i < dim; i += 256) s += xr[i];
    red[threadIdx.x] = s; __syncthreads();
    for (int st = 128; st > 0; st >>= 1) {
        if (threadIdx.x < st) red[threadIdx.x] += red[threadIdx.x + st];
        __syncthreads();
    }
    const float mu = red[0] / dim;
    __syncthreads();
    float vs = 0.0f;
    for (int i = threadIdx.x; i < dim; i += 256) { float d = xr[i] - mu; vs += d * d; }
    red[threadIdx.x] = vs; __syncthreads();
    for (int st = 128; st > 0; st >>= 1) {
        if (threadIdx.x < st) red[threadIdx.x] += red[threadIdx.x + st];
        __syncthreads();
    }
    const float inv = rsqrtf(red[0] / dim + 1e-5f);
    for (int i = threadIdx.x; i < dim; i += 256)
        out[row * dim + i] = (xr[i] - mu) * inv * g[i] + bta[i];
}

// ---------------------------------------------------------------------------
// GLA recurrent scan: one block per (batch, head), 64 threads = DV columns.
// ---------------------------------------------------------------------------
__global__ __launch_bounds__(64)
void gla_scan_kernel(const float* __restrict__ q, const float* __restrict__ k,
                     const float* __restrict__ v, const float* __restrict__ gk,
                     float* __restrict__ o)
{
    const int b   = blockIdx.x / Hh;
    const int h   = blockIdx.x % Hh;
    const int tid = threadIdx.x;
    __shared__ float sq[DKk], sk[DKk], se[DKk];

    float S[DKk];
    #pragma unroll
    for (int i = 0; i < DKk; ++i) S[i] = 0.0f;
    const float scale = 0.17677669529663687f;   // DK^-0.5

    for (int t = 0; t < Nt; ++t) {
        const long long rq = ((long long)(b * Nt + t)) * (Hh * DKk) + h * DKk;
        const long long rv = ((long long)(b * Nt + t)) * (Hh * DVv) + h * DVv;
        if (tid < DKk) {
            sq[tid] = q[rq + tid];
            sk[tid] = k[rq + tid];
            se[tid] = expf(gk[rq + tid]);
        }
        __syncthreads();
        const float vt = v[rv + tid];
        float accv = 0.0f;
        #pragma unroll
        for (int kk = 0; kk < DKk; ++kk) {
            S[kk] = S[kk] * se[kk] + sk[kk] * vt;
            accv += sq[kk] * S[kk];
        }
        o[rv + tid] = accv * scale;
        __syncthreads();
    }
}

// ---------------------------------------------------------------------------
// o = rmsnorm(o, gnorm) * silu(g)
// ---------------------------------------------------------------------------
__global__ __launch_bounds__(64)
void rms_silu_kernel(float* __restrict__ o, const float* __restrict__ g,
                     const float* __restrict__ w)
{
    __shared__ float red[64];
    const long long row = blockIdx.x;
    const float x = o[row * DVv + threadIdx.x];
    red[threadIdx.x] = x * x; __syncthreads();
    for (int st = 32; st > 0; st >>= 1) {
        if (threadIdx.x < st) red[threadIdx.x] += red[threadIdx.x + st];
        __syncthreads();
    }
    const float inv = rsqrtf(red[0] / DVv + 1e-5f);
    const float gv  = g[row * DVv + threadIdx.x];
    const float sil = gv / (1.0f + expf(-gv));
    o[row * DVv + threadIdx.x] = x * inv * w[threadIdx.x] * sil;
}

// ---------------------------------------------------------------------------
// SwiGLU combine: out[r][c] = silu(xx[r][c]) * xx[r][HIN+c]; out has ld=HSW_LD
// ---------------------------------------------------------------------------
__global__ void swiglu_kernel(const float* __restrict__ xx, float* __restrict__ out,
                              long long total)
{
    for (long long i = (long long)blockIdx.x * blockDim.x + threadIdx.x; i < total;
         i += (long long)gridDim.x * blockDim.x) {
        const long long r = i / HINc;
        const int c = (int)(i % HINc);
        const float a  = xx[r * (2 * HINc) + c];
        const float bb = xx[r * (2 * HINc) + HINc + c];
        out[r * HSW_LD + c] = (a / (1.0f + expf(-a))) * bb;
    }
}

// ---------------------------------------------------------------------------
// Softmax over contiguous rows of length 256 (in place)
// ---------------------------------------------------------------------------
__global__ __launch_bounds__(256)
void softmax256_kernel(float* __restrict__ x)
{
    __shared__ float red[256];
    const long long row = blockIdx.x;
    const float v = x[row * 256 + threadIdx.x];
    red[threadIdx.x] = v; __syncthreads();
    for (int st = 128; st > 0; st >>= 1) {
        if (threadIdx.x < st) red[threadIdx.x] = fmaxf(red[threadIdx.x], red[threadIdx.x + st]);
        __syncthreads();
    }
    const float mx = red[0]; __syncthreads();
    const float e = expf(v - mx);
    red[threadIdx.x] = e; __syncthreads();
    for (int st = 128; st > 0; st >>= 1) {
        if (threadIdx.x < st) red[threadIdx.x] += red[threadIdx.x + st];
        __syncthreads();
    }
    x[row * 256 + threadIdx.x] = e / red[0];
}

// ---------------------------------------------------------------------------
// Sinusoidal positions
// ---------------------------------------------------------------------------
__global__ void sinpos_kernel(float* __restrict__ pos)
{
    const int i = blockIdx.x * blockDim.x + threadIdx.x;
    if (i >= Jt * Dm) return;
    const int j = i / Dm, d = i % Dm;
    const int e = d % (Dm / 2);
    const float ex = 2.0f * (float)e / (float)Dm;
    const float theta = (float)j * powf(10000.0f, -ex);
    pos[i] = (d < Dm / 2) ? sinf(theta) : cosf(theta);
}

// ---------------------------------------------------------------------------
// Host-side orchestration
// ---------------------------------------------------------------------------
static void launch_gemm(hipStream_t s, const float* A, const float* Bm,
                        const float* bias, const float* resid, float* C,
                        int M, int N, int K, int lda, int ldb, int ldc,
                        long long sA, long long sB, long long sC, long long sR,
                        int batch, int transB, float alpha, int act)
{
    dim3 grid((N + 15) / 16, ((M + 15) / 16 + 7) / 8, batch);
    if (transB)
        gemm_nt_wmma<<<grid, 256, 0, s>>>(A, Bm, bias, resid, C, M, N, K, lda, ldb,
                                          ldc, sA, sB, sC, sR, alpha, act);
    else
        gemm_nn_wmma<<<grid, 256, 0, s>>>(A, Bm, bias, resid, C, M, N, K, lda, ldb,
                                          ldc, sA, sB, sC, sR, alpha, act);
}

// block-param order: 0 ln1_g 1 ln1_b 2 ln2_g 3 ln2_b 4 wq 5 wk 6 wv 7 wg 8 wo
//                    9 gk_a 10 gk_b 11 gk_bias 12 gnorm 13 w_in 14 b_in 15 w_out 16 b_out
static const long long kBlockParamElems[17] = {
    Dm, Dm, Dm, Dm,
    (long long)Dm * Dm, (long long)Dm * Dm,
    (long long)Dm * 2 * Dm, (long long)Dm * 2 * Dm, (long long)2 * Dm * Dm,
    (long long)Dm * 16, (long long)16 * Dm, Dm, DVv,
    (long long)Dm * 2 * HINc, 2 * HINc, (long long)HINc * Dm, Dm };

struct BlockP { const float* p[17]; };

struct WSBufs {
    float *t, *q, *k, *v, *g, *gk, *t16, *o, *xx, *hsw;
    float *h, *pos, *qc, *kc, *vc, *x1, *a1, *a2;
};

static void run_block(hipStream_t s, const BlockP& P, float* hbuf, const WSBufs& W)
{
    layernorm_kernel<<<BN, 256, 0, s>>>(hbuf, P.p[0], P.p[1], W.t, Dm);
    launch_gemm(s, W.t, P.p[4], nullptr, nullptr, W.q,  BN, Dm,   Dm, Dm, Dm,   Dm,   0,0,0,0, 1, 0, 1.0f, ACT_NONE);
    launch_gemm(s, W.t, P.p[5], nullptr, nullptr, W.k,  BN, Dm,   Dm, Dm, Dm,   Dm,   0,0,0,0, 1, 0, 1.0f, ACT_NONE);
    launch_gemm(s, W.t, P.p[6], nullptr, nullptr, W.v,  BN, 2*Dm, Dm, Dm, 2*Dm, 2*Dm, 0,0,0,0, 1, 0, 1.0f, ACT_NONE);
    launch_gemm(s, W.t, P.p[7], nullptr, nullptr, W.g,  BN, 2*Dm, Dm, Dm, 2*Dm, 2*Dm, 0,0,0,0, 1, 0, 1.0f, ACT_NONE);
    launch_gemm(s, W.t,   P.p[9],  nullptr, nullptr, W.t16, BN, 16, Dm, Dm, 16, 16, 0,0,0,0, 1, 0, 1.0f, ACT_NONE);
    launch_gemm(s, W.t16, P.p[10], P.p[11], nullptr, W.gk,  BN, Dm, 16, 16, Dm, Dm, 0,0,0,0, 1, 0, 1.0f, ACT_LOGSIG_GLN);
    gla_scan_kernel<<<Bc * Hh, 64, 0, s>>>(W.q, W.k, W.v, W.gk, W.o);
    rms_silu_kernel<<<BN * Hh, 64, 0, s>>>(W.o, W.g, P.p[12]);
    launch_gemm(s, W.o, P.p[8], nullptr, hbuf, hbuf, BN, Dm, 2*Dm, 2*Dm, Dm, Dm, 0,0,0,0, 1, 0, 1.0f, ACT_NONE);
    layernorm_kernel<<<BN, 256, 0, s>>>(hbuf, P.p[2], P.p[3], W.t, Dm);
    launch_gemm(s, W.t, P.p[13], P.p[14], nullptr, W.xx, BN, 2*HINc, Dm, Dm, 2*HINc, 2*HINc, 0,0,0,0, 1, 0, 1.0f, ACT_NONE);
    swiglu_kernel<<<1024, 256, 0, s>>>(W.xx, W.hsw, (long long)BN * HINc);
    launch_gemm(s, W.hsw, P.p[15], P.p[16], hbuf, hbuf, BN, Dm, HINc, HSW_LD, Dm, Dm, 0,0,0,0, 1, 0, 1.0f, ACT_NONE);
}

extern "C" void kernel_launch(void* const* d_in, const int* in_sizes, int n_in,
                              void* d_out, int out_size, void* d_ws, size_t ws_size,
                              hipStream_t stream)
{
    (void)in_sizes; (void)n_in; (void)out_size; (void)ws_size;
    const float* x    = (const float*)d_in[0];
    const float* xenc = (const float*)d_in[1];

    float* ws = (float*)d_ws;
    long long off = 0;
    auto take = [&](long long n) { float* p = ws + off; off += n; return p; };
    WSBufs W;
    W.t   = take((long long)BN * Dm);
    W.q   = take((long long)BN * Dm);
    W.k   = take((long long)BN * Dm);
    W.v   = take((long long)BN * 2 * Dm);
    W.g   = take((long long)BN * 2 * Dm);
    W.gk  = take((long long)BN * Dm);
    W.t16 = take((long long)BN * 16);
    W.o   = take((long long)BN * 2 * Dm);
    W.xx  = take((long long)BN * 2 * HINc);
    W.hsw = take((long long)BN * HSW_LD);
    W.h   = take((long long)BN * Dm);
    W.pos = take((long long)Jt * Dm);
    W.qc  = take((long long)BN * Dm);
    W.kc  = take((long long)Bc * Jt * Dm);
    W.vc  = take((long long)Bc * Jt * Dm);
    W.x1  = take((long long)BN * Dm);
    W.a1  = take((long long)Bc * Nt * Jt);
    W.a2  = take((long long)Bc * Nt * Jt);

    BlockP enc0, enc1, dec0, dec1, posb;
    for (int j = 0; j < 17; ++j) {
        enc0.p[j] = (const float*)d_in[4 + j];
        enc1.p[j] = enc0.p[j] + kBlockParamElems[j];
        dec0.p[j] = (const float*)d_in[21 + j];
        dec1.p[j] = dec0.p[j] + kBlockParamElems[j];
        posb.p[j] = (const float*)d_in[50 + j];
    }
    const float* cwq   = (const float*)d_in[38];
    const float* cbq   = (const float*)d_in[39];
    const float* cwk   = (const float*)d_in[40];
    const float* cbk   = (const float*)d_in[41];
    const float* cwv   = (const float*)d_in[42];
    const float* cbv   = (const float*)d_in[43];
    const float* lnq_g = (const float*)d_in[44];
    const float* lnq_b = (const float*)d_in[45];
    const float* lnk_g = (const float*)d_in[46];
    const float* lnk_b = (const float*)d_in[47];
    const float* lnv_g = (const float*)d_in[48];
    const float* lnv_b = (const float*)d_in[49];

    // ---- encoder stack ----
    hipMemcpyAsync(W.h, x, sizeof(float) * BN * Dm, hipMemcpyDeviceToDevice, stream);
    run_block(stream, enc0, W.h, W);
    run_block(stream, enc1, W.h, W);

    // ---- cross attention ----
    const float alphaD = 1.0f / 32.0f;   // D^-0.5
    launch_gemm(stream, W.h, cwq, cbq, nullptr, W.t, BN, Dm, Dm, Dm, Dm, Dm, 0,0,0,0, 1, 0, 1.0f, ACT_NONE);
    layernorm_kernel<<<BN, 256, 0, stream>>>(W.t, lnq_g, lnq_b, W.qc, Dm);
    launch_gemm(stream, xenc, cwk, cbk, nullptr, W.t, Bc * Jt, Dm, DCc, DCc, Dm, Dm, 0,0,0,0, 1, 0, 1.0f, ACT_NONE);
    layernorm_kernel<<<Bc * Jt, 256, 0, stream>>>(W.t, lnk_g, lnk_b, W.kc, Dm);
    launch_gemm(stream, xenc, cwv, cbv, nullptr, W.t, Bc * Jt, Dm, DCc, DCc, Dm, Dm, 0,0,0,0, 1, 0, 1.0f, ACT_NONE);
    layernorm_kernel<<<Bc * Jt, 256, 0, stream>>>(W.t, lnv_g, lnv_b, W.vc, Dm);
    sinpos_kernel<<<(Jt * Dm + 255) / 256, 256, 0, stream>>>(W.pos);
    // a1 = softmax(q k^T / sqrt(D))
    launch_gemm(stream, W.qc, W.kc, nullptr, nullptr, W.a1, Nt, Jt, Dm, Dm, Dm, Jt,
                (long long)Nt * Dm, (long long)Jt * Dm, (long long)Nt * Jt, 0,
                Bc, 1, alphaD, ACT_NONE);
    softmax256_kernel<<<Bc * Nt, 256, 0, stream>>>(W.a1);
    // x1 = a1 @ pos
    launch_gemm(stream, W.a1, W.pos, nullptr, nullptr, W.x1, Nt, Dm, Jt, Jt, Dm, Dm,
                (long long)Nt * Jt, 0, (long long)Nt * Dm, 0, Bc, 0, 1.0f, ACT_NONE);
    run_block(stream, posb, W.x1, W);
    // a2 = softmax(x1 pos^T / sqrt(D))
    launch_gemm(stream, W.x1, W.pos, nullptr, nullptr, W.a2, Nt, Jt, Dm, Dm, Dm, Jt,
                (long long)Nt * Dm, 0, (long long)Nt * Jt, 0, Bc, 1, alphaD, ACT_NONE);
    softmax256_kernel<<<Bc * Nt, 256, 0, stream>>>(W.a2);
    // h = a2 @ v + h
    launch_gemm(stream, W.a2, W.vc, nullptr, W.h, W.h, Nt, Dm, Jt, Jt, Dm, Dm,
                (long long)Nt * Jt, (long long)Jt * Dm, (long long)Nt * Dm,
                (long long)Nt * Dm, Bc, 0, 1.0f, ACT_NONE);

    // ---- decoder stack ----
    run_block(stream, dec0, W.h, W);
    run_block(stream, dec1, W.h, W);

    // ---- outputs: y (B,N,D) then att (B,2,N,J) ----
    float* out = (float*)d_out;
    hipMemcpyAsync(out, W.h, sizeof(float) * BN * Dm, hipMemcpyDeviceToDevice, stream);
    float* att = out + (long long)BN * Dm;
    const long long sl = (long long)Nt * Jt;
    for (int b = 0; b < Bc; ++b) {
        hipMemcpyAsync(att + b * 2 * sl,      W.a1 + b * sl, sizeof(float) * sl,
                       hipMemcpyDeviceToDevice, stream);
        hipMemcpyAsync(att + b * 2 * sl + sl, W.a2 + b * sl, sizeof(float) * sl,
                       hipMemcpyDeviceToDevice, stream);
    }
}
